// FP8LinearBase_686
// MI455X (gfx1250) — 8-bit matrix operand order repaired and checked by reading, compiled, not run on hardware
//
#include <hip/hip_runtime.h>
#include <hip/hip_bf16.h>


typedef __attribute__((ext_vector_type(4)))  int   v4i;
typedef __attribute__((ext_vector_type(16))) int   v16i;
typedef __attribute__((ext_vector_type(8)))  float v8f;

#define M_DIM 8192
#define K_DIM 4096
#define N_DIM 4096
#define KT    (K_DIM / 128)
#define FP8_MAXF 448.0f
#define STAGE_BYTES (16 * 2048)


__device__ __forceinline__ unsigned f32_to_e4m3_sw(float xf) {
  unsigned u = __float_as_uint(xf);
  unsigned sign = (u >> 24) & 0x80u;
  u &= 0x7FFFFFFFu;
  if (u > 0x43E00000u) u = 0x43E00000u;
  int e = (int)(u >> 23);
  unsigned man = u & 0x7FFFFFu;
  int et = e - 120;
  unsigned r;
  if (et >= 1) {
    unsigned rnd = man + 0x7FFFFu + ((man >> 20) & 1u);
    unsigned m3 = rnd >> 20;
    if (m3 & 8u) { m3 = 0u; ++et; }
    r = ((unsigned)et << 3) | m3;
    if (r > 0x7Eu) r = 0x7Eu;
  } else {
    int sh = 21 - et;
    if (sh > 24) {
      r = 0u;
    } else {
      unsigned full = 0x800000u | man;
      unsigned shifted = full >> sh;
      unsigned rem = full & ((1u << sh) - 1u);
      unsigned half = 1u << (sh - 1);
      if (rem > half || (rem == half && (shifted & 1u))) ++shifted;
      r = shifted;
    }
  }
  return sign | r;
}

__device__ __forceinline__ unsigned quant_pack4(float f0, float f1, float f2, float f3) {
  f0 = fminf(FP8_MAXF, fmaxf(-FP8_MAXF, f0));
  f1 = fminf(FP8_MAXF, fmaxf(-FP8_MAXF, f1));
  f2 = fminf(FP8_MAXF, fmaxf(-FP8_MAXF, f2));
  f3 = fminf(FP8_MAXF, fmaxf(-FP8_MAXF, f3));
#if __has_builtin(__builtin_amdgcn_cvt_pk_fp8_f32)
  int lo = __builtin_amdgcn_cvt_pk_fp8_f32(f0, f1, 0,  false);
  int hi = __builtin_amdgcn_cvt_pk_fp8_f32(f2, f3, lo, true);
  return (unsigned)hi;
#else
  return f32_to_e4m3_sw(f0)        | (f32_to_e4m3_sw(f1) << 8) |
        (f32_to_e4m3_sw(f2) << 16) | (f32_to_e4m3_sw(f3) << 24);
#endif
}


__global__ __launch_bounds__(256) void quant_pack_a(
    const float* __restrict__ x, const float* __restrict__ iscale,
    unsigned* __restrict__ apack) {
  const unsigned t = blockIdx.x * blockDim.x + threadIdx.x;
  const unsigned tile = t >> 9;
  const unsigned q    = t & 511u;
  const unsigned tm   = tile >> 5;
  const unsigned tk   = tile & 31u;
  const unsigned lane = q >> 4;
  const unsigned v    = q & 15u;

  const unsigned m  = tm * 16u + (lane & 15u);
  const unsigned kb = ((v >> 2) * 32u) + ((lane >> 4) * 16u) +
                      ((v & 3u) * 4u);
  const unsigned k  = tk * 128u + kb;

  const float4 f = *reinterpret_cast<const float4*>(x + (size_t)m * K_DIM + k);
  const float inv = 1.0f / iscale[0];
  const float g0 = __bfloat162float(__float2bfloat16(f.x)) * inv;
  const float g1 = __bfloat162float(__float2bfloat16(f.y)) * inv;
  const float g2 = __bfloat162float(__float2bfloat16(f.z)) * inv;
  const float g3 = __bfloat162float(__float2bfloat16(f.w)) * inv;
  apack[t] = quant_pack4(g0, g1, g2, g3);
}


__global__ __launch_bounds__(256) void pack_b(
    const unsigned char* __restrict__ w, unsigned* __restrict__ bpack) {
  const unsigned t = blockIdx.x * blockDim.x + threadIdx.x;
  const unsigned tile = t >> 9;
  const unsigned q    = t & 511u;
  const unsigned tn   = tile >> 5;
  const unsigned tk   = tile & 31u;
  const unsigned lane = q >> 4;
  const unsigned v    = q & 15u;

  const unsigned n = tn * 16u + (lane & 15u);
  const unsigned k = tk * 128u + ((v >> 2) * 32u) + ((lane >> 4) * 16u) +
                     ((v & 3u) * 4u);

  const size_t base = (size_t)k * N_DIM + n;
  const unsigned b0 = w[base];
  const unsigned b1 = w[base + (size_t)N_DIM];
  const unsigned b2 = w[base + (size_t)2 * N_DIM];
  const unsigned b3 = w[base + (size_t)3 * N_DIM];
  bpack[t] = b0 | (b1 << 8) | (b2 << 16) | (b3 << 24);
}


#define ASYNC_CP16(OFF_LIT)                                                   \
  asm volatile("global_load_async_to_lds_b128 %0, %1, off offset:" #OFF_LIT  \
               :: "v"(ldsAddr), "v"(src) : "memory")

__device__ __forceinline__ void issue_stage(unsigned ldsAddr,
                                            const unsigned char* src) {
  ASYNC_CP16(0);   ASYNC_CP16(16);  ASYNC_CP16(32);  ASYNC_CP16(48);
  ASYNC_CP16(64);  ASYNC_CP16(80);  ASYNC_CP16(96);  ASYNC_CP16(112);
}

__global__ __launch_bounds__(256) void fp8_gemm(
    const unsigned char* __restrict__ apack, const unsigned char* __restrict__ bpack,
    const float* __restrict__ wscale, const float* __restrict__ iscale,
    const float* __restrict__ bias, float* __restrict__ out) {
  __shared__ __align__(128) unsigned char smem[3][STAGE_BYTES];

  const int tid  = threadIdx.x;
  const int lane = tid & 31;
  const int wave = tid >> 5;
  const int bm   = blockIdx.x & 63;
  const int bn   = blockIdx.x >> 6;
  const int wm   = wave & 1;
  const int wn   = wave >> 1;

  const int chunk = tid >> 4;
  const int coff  = (tid & 15) * 128;
  const unsigned char* gsrc =
      (chunk < 8)
          ? apack + ((size_t)(bm * 8 + chunk) * KT) * 2048 + coff
          : bpack + ((size_t)(bn * 8 + (chunk - 8)) * KT) * 2048 + coff;
  const unsigned lds0 = (unsigned)(uintptr_t)(&smem[0][0] + chunk * 2048 + coff);

  v8f acc[4][2] = {};

  issue_stage(lds0, gsrc);
  issue_stage(lds0 + STAGE_BYTES, gsrc + 2048);

  int cur = 0;
  for (int tk = 0; tk < KT; ++tk) {
    if (tk + 1 < KT) asm volatile("s_wait_asynccnt 0x8" ::: "memory");
    else             asm volatile("s_wait_asynccnt 0x0" ::: "memory");
    __syncthreads();

    if (tk + 2 < KT) {
      int pre = cur + 2; if (pre >= 3) pre -= 3;
      issue_stage(lds0 + (unsigned)pre * STAGE_BYTES,
                  gsrc + (size_t)(tk + 2) * 2048);
    }

    const unsigned char* sa = &smem[0][0] + cur * STAGE_BYTES;
    v16i a[4], b[2];
#pragma unroll
    for (int i = 0; i < 4; ++i)
      a[i] = *reinterpret_cast<const v16i*>(sa + (wm * 4 + i) * 2048 + lane * 64);
#pragma unroll
    for (int j = 0; j < 2; ++j)
      b[j] = *reinterpret_cast<const v16i*>(sa + (8 + wn * 2 + j) * 2048 + lane * 64);

#pragma unroll
    for (int i = 0; i < 4; ++i)
#pragma unroll
      for (int j = 0; j < 2; ++j)
        acc[i][j] = __builtin_amdgcn_wmma_f32_16x16x128_fp8_fp8(
            a[i], b[j], (short)0, acc[i][j], false, false);

    if (++cur == 3) cur = 0;
  }

  const float is = iscale[0];
  const int nloc = lane & 15;
  const int rowHalf = (lane >> 4) * 8;
  const int tm0 = bm * 8 + wm * 4;
  const int tn0 = bn * 8 + wn * 2;
#pragma unroll
  for (int j = 0; j < 2; ++j) {
    const int n = (tn0 + j) * 16 + nloc;
    const float s  = is * wscale[n];
    const float bb = bias[n];
#pragma unroll
    for (int i = 0; i < 4; ++i) {
      const int m0 = (tm0 + i) * 16 + rowHalf;
      float* po = out + (size_t)m0 * N_DIM + n;
#pragma unroll
      for (int r = 0; r < 8; ++r)
        po[(size_t)r * N_DIM] = acc[i][j][r] * s + bb;
    }
  }
}


extern "C" void kernel_launch(void* const* d_in, const int* in_sizes, int n_in,
                              void* d_out, int out_size, void* d_ws, size_t ws_size,
                              hipStream_t stream) {
  const float*         x      = (const float*)d_in[0];
  const unsigned char* qw     = (const unsigned char*)d_in[1];
  const float*         wscale = (const float*)d_in[2];
  const float*         iscale = (const float*)d_in[3];
  const float*         bias   = (const float*)d_in[4];
  float*               out    = (float*)d_out;

  unsigned char* apack = (unsigned char*)d_ws;
  unsigned char* bpack = (unsigned char*)d_ws + (size_t)M_DIM * K_DIM;

  quant_pack_a<<<(M_DIM * K_DIM / 4) / 256, 256, 0, stream>>>(
      x, iscale, (unsigned*)apack);
  pack_b<<<(K_DIM * N_DIM / 4) / 256, 256, 0, stream>>>(qw, (unsigned*)bpack);

  fp8_gemm<<<(M_DIM / 128) * (N_DIM / 128), 256, 0, stream>>>(
      apack, bpack, wscale, iscale, bias, out);
}
